// ViscoplasticMaterialModel_26972394619208
// MI455X (gfx1250) — matrix operand order repaired and checked by reading, not compiled, not run on hardware
//
#include <hip/hip_runtime.h>
#include <hip/hip_bf16.h>
#include <stdint.h>

// ---------------------------------------------------------------------------
// ViscoplasticMaterialModel for MI455X (gfx1250, wave32, WMMA bf16)
//   Phase 1: FNO encoders (direct 16-mode DFT)        -> m1, m2   [B,128]
//   Phase 2: time-invariant PICNN context (WMMA GEMM) -> c1e..c2d [B,128]
//   Phase 3: T=64 serial scan; per step 4x (16x128x128) bf16 WMMA GEMMs,
//            wave-parallel gradient reduction via LDS float atomics.
// Workspace use: ~29.5 MB.
// ---------------------------------------------------------------------------

#define B_TOT   8192
#define T_STEPS 64
#define DT_STEP 1e-3f

typedef __bf16 bf16;
typedef __attribute__((ext_vector_type(16))) __bf16 v16bf;
typedef __attribute__((ext_vector_type(8)))  float  v8f;

__device__ __forceinline__ float softplusf(float x) {
  if (x > 20.f) return x;
  if (x < -20.f) return __expf(x);
  return log1pf(__expf(x));
}
__device__ __forceinline__ float sigm(float x) { return 1.f / (1.f + __expf(-x)); }
__device__ __forceinline__ float geluf(float x) {
  float t = tanhf(0.7978845608028654f * (x + 0.044715f * x * x * x));
  return 0.5f * x * (1.f + t);
}

// A-fragment: 16x32 bf16, row-major source with leading dim ldk (bf16 elems).
// ISA layout: lanes 0-15 -> K[0..7] in v0..3, K[16..23] in v4..7 (row = lane);
//             lanes 16-31 -> K[8..15], K[24..31] (row = lane-16).
__device__ __forceinline__ v16bf load_a_frag(const bf16* base, int ldk, int k0, int lane) {
  int row = lane & 15, half = (lane >> 4) & 1;
  const bf16* p = base + row * ldk + k0 + half * 8;
  union { v16bf v; uint4 q[2]; } u;
  u.q[0] = *(const uint4*)(p);
  u.q[1] = *(const uint4*)(p + 16);
  return u.v;
}
// B-fragment for out = act @ W.T (B[k][n] = W[n0+n][k0+k], W row-major, ld=ldk):
// lanes 0-15 hold K[0..7],K[16..23] of row (n0+lane), lanes 16-31 hold K[8..15],K[24..31].
__device__ __forceinline__ v16bf load_b_frag(const bf16* W, int ldk, int n0, int k0, int lane) {
  int col = lane & 15, half = (lane >> 4) & 1;
  const bf16* p = W + (size_t)(n0 + col) * ldk + k0 + half * 8;
  union { v16bf v; uint4 q[2]; } u;
  u.q[0] = *(const uint4*)(p);
  u.q[1] = *(const uint4*)(p + 16);
  return u.v;
}
__device__ __forceinline__ v8f wmma_bf16(v16bf a, v16bf b, v8f c) {
  return __builtin_amdgcn_wmma_f32_16x16x32_bf16(false, a, false, b, (short)0, c, false, false);
}

// ---------------------------------------------------------------------------
// FNO-style 1d functional encoder. One block (128 thr) per batch element.
// ---------------------------------------------------------------------------
__global__ __launch_bounds__(128) void fnf_kernel(
    const float* __restrict__ x1, const float* __restrict__ x2, int dimIn,
    const float* __restrict__ P,   const float* __restrict__ WcRe,
    const float* __restrict__ WcIm,const float* __restrict__ Wp,
    const float* __restrict__ Wf1, const float* __restrict__ bf1,
    const float* __restrict__ Wf2, const float* __restrict__ bf2,
    float* __restrict__ mOut)
{
  __shared__ float sv[2][32][128];
  __shared__ float cosT[16 * 128], sinT[16 * 128];
  __shared__ float vfr[32 * 16], vfi[32 * 16];
  __shared__ float hre[32 * 16], him[32 * 16];
  __shared__ float pooled[32], hbuf[64];

  const int tid = threadIdx.x;
  const int b = blockIdx.x;
  const float w0 = 6.2831853071795864f / 128.f;

  for (int idx = tid; idx < 16 * 128; idx += 128) {
    int m = idx >> 7, nn = idx & 127;
    int ph = (m * nn) & 127;                  // exact phase reduction
    float th = w0 * (float)ph;
    cosT[idx] = __cosf(th); sinT[idx] = __sinf(th);
  }
  for (int idx = tid; idx < 32 * 128; idx += 128) {
    int w = idx >> 7, nn = idx & 127;
    float acc = P[w * dimIn + 0] * x1[(size_t)b * 128 + nn];
    if (dimIn == 2) acc += P[w * dimIn + 1] * x2[(size_t)b * 128 + nn];
    sv[0][w][nn] = acc;
  }
  __syncthreads();

  int cur = 0;
  for (int l = 0; l < 3; ++l) {
    // forward DFT, first 16 modes only
    for (int idx = tid; idx < 512; idx += 128) {
      int i = idx >> 4, m = idx & 15;
      float ar = 0.f, ai = 0.f;
      for (int nn = 0; nn < 128; ++nn) {
        float vv = sv[cur][i][nn];
        ar += vv * cosT[m * 128 + nn];
        ai -= vv * sinT[m * 128 + nn];
      }
      vfr[idx] = ar; vfi[idx] = ai;
    }
    __syncthreads();
    // per-mode 32x32 complex channel mix
    for (int idx = tid; idx < 512; idx += 128) {
      int o = idx >> 4, m = idx & 15;
      float ar = 0.f, ai = 0.f;
      for (int i = 0; i < 32; ++i) {
        float wr = WcRe[((l * 32 + i) * 32 + o) * 16 + m];
        float wi = WcIm[((l * 32 + i) * 32 + o) * 16 + m];
        float xr = vfr[i * 16 + m], xim = vfi[i * 16 + m];
        ar += xr * wr - xim * wi;
        ai += xr * wi + xim * wr;
      }
      hre[idx] = ar; him[idx] = ai;
    }
    __syncthreads();
    // iDFT + pointwise Wp + gelu
    for (int idx = tid; idx < 32 * 128; idx += 128) {
      int o = idx >> 7, nn = idx & 127;
      float s = hre[o * 16 + 0];
      #pragma unroll
      for (int m = 1; m < 16; ++m)
        s += 2.f * (hre[o * 16 + m] * cosT[m * 128 + nn] -
                    him[o * 16 + m] * sinT[m * 128 + nn]);
      s *= (1.f / 128.f);
      float pw = 0.f;
      #pragma unroll
      for (int i = 0; i < 32; ++i) pw += Wp[(l * 32 + o) * 32 + i] * sv[cur][i][nn];
      float out = s + pw;
      sv[1 - cur][o][nn] = (l < 2) ? geluf(out) : out;
    }
    __syncthreads();
    cur = 1 - cur;
  }
  if (tid < 32) {
    float s = 0.f;
    for (int nn = 0; nn < 128; ++nn) s += sv[cur][tid][nn];
    pooled[tid] = s * (1.f / 128.f);
  }
  __syncthreads();
  if (tid < 64) {
    float s = bf1[tid];
    #pragma unroll
    for (int i = 0; i < 32; ++i) s += Wf1[tid * 32 + i] * pooled[i];
    hbuf[tid] = geluf(s);
  }
  __syncthreads();
  {
    float s = bf2[tid];
    #pragma unroll
    for (int k = 0; k < 64; ++k) s += Wf2[tid * 64 + k] * hbuf[k];
    mOut[(size_t)b * 128 + tid] = s;
  }
}

// ---------------------------------------------------------------------------
// Weight precompute: bf16 softplus(Wz2) (+transpose), bf16 copies of the six
// context matrices, softplus(wz_out). One pass, done once per call.
// ---------------------------------------------------------------------------
__global__ __launch_bounds__(256) void wconv_kernel(
    const float* __restrict__ Wz2e, const float* __restrict__ Wz2d,
    const float* __restrict__ wzoute, const float* __restrict__ wzoutd,
    const float* __restrict__ WuE, const float* __restrict__ Wuy1E,
    const float* __restrict__ Wuz2E,
    const float* __restrict__ WuD, const float* __restrict__ Wuy1D,
    const float* __restrict__ Wuz2D,
    bf16* __restrict__ WspE, bf16* __restrict__ WspET,
    bf16* __restrict__ WspD, bf16* __restrict__ WspDT,
    bf16* __restrict__ WuEb, bf16* __restrict__ Wuy1Eb, bf16* __restrict__ Wuz2Eb,
    bf16* __restrict__ WuDb, bf16* __restrict__ Wuy1Db, bf16* __restrict__ Wuz2Db,
    float* __restrict__ wzspe, float* __restrict__ wzspd)
{
  int idx = blockIdx.x * blockDim.x + threadIdx.x;
  if (idx < 128 * 128) {
    int n = idx >> 7, k = idx & 127;
    float ve = softplusf(Wz2e[idx]);
    WspE[idx] = (bf16)ve; WspET[k * 128 + n] = (bf16)ve;
    float vd = softplusf(Wz2d[idx]);
    WspD[idx] = (bf16)vd; WspDT[k * 128 + n] = (bf16)vd;
    WuEb[idx]   = (bf16)WuE[idx];
    Wuy1Eb[idx] = (bf16)Wuy1E[idx];
    Wuz2Eb[idx] = (bf16)Wuz2E[idx];
    WuDb[idx]   = (bf16)WuD[idx];
    Wuy1Db[idx] = (bf16)Wuy1D[idx];
    Wuz2Db[idx] = (bf16)Wuz2D[idx];
  }
  if (idx < 128) { wzspe[idx] = softplusf(wzoute[idx]); wzspd[idx] = softplusf(wzoutd[idx]); }
}

// ---------------------------------------------------------------------------
// out[B,128] = act( X[B,128] @ Wb[128,128].T (+bias) ), bf16 WMMA, f32 acc.
// 16 rows/block, 8 waves; B-fragments straight from global bf16 (L2-hot).
// ---------------------------------------------------------------------------
__global__ __launch_bounds__(256) void gemm128_kernel(
    const float* __restrict__ X, const bf16* __restrict__ Wb,
    const float* __restrict__ bias, float* __restrict__ out, int act)
{
  __shared__ bf16 sX[16 * 128];
  const int tid = threadIdx.x;
  const size_t row0 = (size_t)blockIdx.x * 16;
  for (int idx = tid; idx < 16 * 128; idx += 256)
    sX[idx] = (bf16)X[row0 * 128 + idx];
  __syncthreads();

  const int wave = tid >> 5, lane = tid & 31;
  const int n0 = wave * 16;
  const int ncol = n0 + (lane & 15);
  const int mh = (lane >> 4) * 8;
  v8f acc = {0.f, 0.f, 0.f, 0.f, 0.f, 0.f, 0.f, 0.f};
  #pragma unroll
  for (int kc = 0; kc < 4; ++kc) {
    v16bf a = load_a_frag(sX, 128, kc * 32, lane);
    v16bf b = load_b_frag(Wb, 128, n0, kc * 32, lane);
    acc = wmma_bf16(a, b, acc);
  }
  float bs = bias ? bias[ncol] : 0.f;
  #pragma unroll
  for (int r = 0; r < 8; ++r) {
    float v = acc[r] + bs;
    if (act == 1) v = softplusf(v);
    out[(row0 + r + mh) * 128 + ncol] = v;
  }
}

// ---------------------------------------------------------------------------
// One PICNN gradient pass over a 16-row tile (forward + analytic backward).
// Leaves grad[m][j] (j<JDIM) in sGrad (ld 12). 8 WMMAs per wave; gradient
// assembly is wave-parallel over 16-column slices with LDS float atomics.
// ---------------------------------------------------------------------------
template <int JDIM>
__device__ __forceinline__ void picnn_grad(
    int tid, int wave, int lane, size_t rowbase,
    const float* __restrict__ c1g, const float* __restrict__ c2g,
    const float* sWy1, const float* sWy2, const float* swz, const float* swy,
    const v16bf* Bf, const v16bf* BfT, const float* sYin,
    bf16* sAct, float* sS1, float* sG2, float* sGrad)
{
  // seed grads with wy_out (sGrad is free at entry; consumed before z1 barrier)
  for (int idx = tid; idx < 16 * JDIM; idx += 256) {
    int m = idx / JDIM, j = idx - m * JDIM;
    sGrad[m * 12 + j] = swy[j];
  }
  // z1pre = y@Wy1.T + c1 ; stash softplus (bf16 A operand) and sigmoid
  for (int idx = tid; idx < 16 * 128; idx += 256) {
    int m = idx >> 7, nn = idx & 127;
    float s = c1g[rowbase + (size_t)m * 128 + nn];
    #pragma unroll
    for (int j = 0; j < JDIM; ++j) s += sYin[m * 12 + j] * sWy1[nn * JDIM + j];
    sAct[idx] = (bf16)softplusf(s);
    sS1[idx] = sigm(s);
  }
  __syncthreads();

  const int n0 = wave * 16;
  const int ncol = n0 + (lane & 15);
  const int mh = (lane >> 4) * 8;

  // z2pre = z1 @ Wsp.T + y@Wy2.T + c2  (acc seeded with the small terms)
  v8f acc;
  #pragma unroll
  for (int r = 0; r < 8; ++r) {
    int m = r + mh;
    float s = c2g[rowbase + (size_t)m * 128 + ncol];
    #pragma unroll
    for (int j = 0; j < JDIM; ++j) s += sYin[m * 12 + j] * sWy2[ncol * JDIM + j];
    acc[r] = s;
  }
  #pragma unroll
  for (int kc = 0; kc < 4; ++kc) {
    v16bf a = load_a_frag(sAct, 128, kc * 32, lane);
    acc = wmma_bf16(a, Bf[kc], acc);
  }
  __syncthreads();                 // all waves finished reading z1 from sAct
  {
    float wn = swz[ncol];
    #pragma unroll
    for (int r = 0; r < 8; ++r) {  // g2 = sigmoid(z2pre) * sp(wz_out)
      int m = r + mh;
      float g = sigm(acc[r]) * wn;
      sAct[m * 128 + ncol] = (bf16)g;
      sG2[m * 128 + ncol] = g;
    }
  }
  __syncthreads();

  // g1 = sigmoid(z1pre) * (g2 @ Wsp)   (contract over n -> use transposed W)
  {
    v8f a2 = {0.f, 0.f, 0.f, 0.f, 0.f, 0.f, 0.f, 0.f};
    #pragma unroll
    for (int kc = 0; kc < 4; ++kc) {
      v16bf a = load_a_frag(sAct, 128, kc * 32, lane);
      a2 = wmma_bf16(a, BfT[kc], a2);
    }
    #pragma unroll
    for (int r = 0; r < 8; ++r) {
      int m = r + mh;
      sS1[m * 128 + ncol] *= a2[r];   // in-place: sS1 now holds g1
    }
  }
  __syncthreads();

  // grad_y[m][j] += sum_n g2*Wy2 + sum_k g1*Wy1 : each wave reduces its own
  // 16-column slice, accumulates with ds_add_f32 atomics.
  for (int p = lane; p < 16 * JDIM; p += 32) {
    int m = p / JDIM, j = p - m * JDIM;
    float s = 0.f;
    #pragma unroll
    for (int k = 0; k < 16; ++k) {
      int nn = n0 + k;
      s += sG2[m * 128 + nn] * sWy2[nn * JDIM + j] +
           sS1[m * 128 + nn] * sWy1[nn * JDIM + j];
    }
    atomicAdd(&sGrad[m * 12 + j], s);
  }
  __syncthreads();
}

// ---------------------------------------------------------------------------
// Serial scan over T: 16 batch rows/block, xi kept in LDS, weights in VGPRs.
// ---------------------------------------------------------------------------
__global__ __launch_bounds__(256) void timeloop_kernel(
    const float* __restrict__ e,
    const float* __restrict__ c1e, const float* __restrict__ c2e,
    const float* __restrict__ c1d, const float* __restrict__ c2d,
    const bf16* __restrict__ WspE, const bf16* __restrict__ WspET,
    const bf16* __restrict__ WspD, const bf16* __restrict__ WspDT,
    const float* __restrict__ wzspe, const float* __restrict__ wzspd,
    const float* __restrict__ Wy1E, const float* __restrict__ Wy2E,
    const float* __restrict__ wyoutE,
    const float* __restrict__ Wy1D, const float* __restrict__ Wy2D,
    const float* __restrict__ wyoutD,
    float* __restrict__ stress_out, float* __restrict__ xis_out)
{
  __shared__ bf16  sAct[16 * 128];
  __shared__ float sS1[16 * 128];
  __shared__ float sG2[16 * 128];
  __shared__ float sY[16 * 12];
  __shared__ float sGrad[16 * 12];
  __shared__ float sXi[16 * 8];
  __shared__ float sE[16 * 64];
  __shared__ float sWy1e[128 * 9], sWy2e[128 * 9];
  __shared__ float sWy1d[128 * 8], sWy2d[128 * 8];
  __shared__ float swz_e[128], swz_d[128];
  __shared__ float swy_e[9], swy_d[8];

  const int tid = threadIdx.x;
  const int wave = tid >> 5, lane = tid & 31;
  const int b0 = blockIdx.x * 16;
  const size_t rowbase = (size_t)b0 * 128;

  for (int idx = tid; idx < 128 * 9; idx += 256) { sWy1e[idx] = Wy1E[idx]; sWy2e[idx] = Wy2E[idx]; }
  for (int idx = tid; idx < 128 * 8; idx += 256) { sWy1d[idx] = Wy1D[idx]; sWy2d[idx] = Wy2D[idx]; }
  if (tid < 128) { swz_e[tid] = wzspe[tid]; swz_d[tid] = wzspd[tid]; }
  if (tid < 9) swy_e[tid] = wyoutE[tid];
  if (tid < 8) swy_d[tid] = wyoutD[tid];
  for (int idx = tid; idx < 16 * 64; idx += 256)
    sE[idx] = e[(size_t)(b0 + (idx >> 6)) * 64 + (idx & 63)];
  for (int idx = tid; idx < 16 * 8; idx += 256) sXi[idx] = 0.f;

  // Weight B-fragments for this wave's 16 output columns, held in VGPRs.
  v16bf BE[4], BET[4], BD[4], BDT[4];
  {
    int n0 = wave * 16;
    #pragma unroll
    for (int kc = 0; kc < 4; ++kc) {
      BE[kc]  = load_b_frag(WspE,  128, n0, kc * 32, lane);
      BET[kc] = load_b_frag(WspET, 128, n0, kc * 32, lane);
      BD[kc]  = load_b_frag(WspD,  128, n0, kc * 32, lane);
      BDT[kc] = load_b_frag(WspDT, 128, n0, kc * 32, lane);
    }
  }
  __syncthreads();

  for (int t = 0; t < T_STEPS; ++t) {
    // emit xi (pre-update), build y = [e_t, xi]
    for (int idx = tid; idx < 16 * 8; idx += 256) {
      int m = idx >> 3, i = idx & 7;
      float x = sXi[idx];
      xis_out[((size_t)(b0 + m) * 64 + t) * 8 + i] = x;
      sY[m * 12 + 1 + i] = x;
    }
    if (tid < 16) sY[tid * 12 + 0] = sE[tid * 64 + t];
    __syncthreads();

    // energy PICNN: grads wrt (e_t, xi)
    picnn_grad<9>(tid, wave, lane, rowbase, c1e, c2e,
                  sWy1e, sWy2e, swz_e, swy_e, BE, BET, sY,
                  sAct, sS1, sG2, sGrad);
    // stress = d/de ; q = -d/dxi
    for (int idx = tid; idx < 16 * 9; idx += 256) {
      int m = idx / 9, j = idx - m * 9;
      float g = sGrad[m * 12 + j];
      if (j == 0) stress_out[(size_t)(b0 + m) * 64 + t] = g;
      else        sY[m * 12 + (j - 1)] = -g;
    }
    __syncthreads();

    // dissipation PICNN: grad wrt q; explicit Euler update of xi
    picnn_grad<8>(tid, wave, lane, rowbase, c1d, c2d,
                  sWy1d, sWy2d, swz_d, swy_d, BD, BDT, sY,
                  sAct, sS1, sG2, sGrad);
    for (int idx = tid; idx < 16 * 8; idx += 256) {
      int m = idx >> 3, i = idx & 7;
      sXi[idx] += DT_STEP * sGrad[m * 12 + i];
    }
    __syncthreads();
  }
}

// ---------------------------------------------------------------------------
extern "C" void kernel_launch(void* const* d_in, const int* in_sizes, int n_in,
                              void* d_out, int out_size, void* d_ws, size_t ws_size,
                              hipStream_t stream)
{
  (void)in_sizes; (void)n_in; (void)out_size; (void)ws_size;
  const float* e      = (const float*)d_in[0];
  const float* E      = (const float*)d_in[1];
  const float* Yg     = (const float*)d_in[2];
  const float* edot0  = (const float*)d_in[3];
  // d_in[4] = n (unused by the reference forward)
  const float* WuE    = (const float*)d_in[5];
  const float* Wy1E   = (const float*)d_in[6];
  const float* Wuy1E  = (const float*)d_in[7];
  const float* Wz2E   = (const float*)d_in[8];
  const float* Wy2E   = (const float*)d_in[9];
  const float* Wuz2E  = (const float*)d_in[10];
  const float* b2E    = (const float*)d_in[11];
  const float* wzoutE = (const float*)d_in[12];
  const float* wyoutE = (const float*)d_in[13];
  const float* WuD    = (const float*)d_in[14];
  const float* Wy1D   = (const float*)d_in[15];
  const float* Wuy1D  = (const float*)d_in[16];
  const float* Wz2D   = (const float*)d_in[17];
  const float* Wy2D   = (const float*)d_in[18];
  const float* Wuz2D  = (const float*)d_in[19];
  const float* b2D    = (const float*)d_in[20];
  const float* wzoutD = (const float*)d_in[21];
  const float* wyoutD = (const float*)d_in[22];
  const float* P1     = (const float*)d_in[23];
  const float* WcRe1  = (const float*)d_in[24];
  const float* WcIm1  = (const float*)d_in[25];
  const float* Wp1    = (const float*)d_in[26];
  const float* Wf1_1  = (const float*)d_in[27];
  const float* bf1_1  = (const float*)d_in[28];
  const float* Wf2_1  = (const float*)d_in[29];
  const float* bf2_1  = (const float*)d_in[30];
  const float* P2     = (const float*)d_in[31];
  const float* WcRe2  = (const float*)d_in[32];
  const float* WcIm2  = (const float*)d_in[33];
  const float* Wp2    = (const float*)d_in[34];
  const float* Wf1_2  = (const float*)d_in[35];
  const float* bf1_2  = (const float*)d_in[36];
  const float* Wf2_2  = (const float*)d_in[37];
  const float* bf2_2  = (const float*)d_in[38];

  char* w = (char*)d_ws;
  auto alloc = [&](size_t bytes) -> void* {
    void* p = (void*)w;
    w += (bytes + 255) & ~(size_t)255;
    return p;
  };
  const size_t BV = (size_t)B_TOT * 128 * sizeof(float);
  const size_t WB = 128 * 128 * sizeof(bf16);
  float* m1   = (float*)alloc(BV);
  float* m2   = (float*)alloc(BV);
  float* c1e  = (float*)alloc(BV);
  float* c2e  = (float*)alloc(BV);
  float* c1d  = (float*)alloc(BV);
  float* c2d  = (float*)alloc(BV);
  float* utmp = (float*)alloc(BV);
  bf16* WspE   = (bf16*)alloc(WB);
  bf16* WspET  = (bf16*)alloc(WB);
  bf16* WspD   = (bf16*)alloc(WB);
  bf16* WspDT  = (bf16*)alloc(WB);
  bf16* WuEb   = (bf16*)alloc(WB);
  bf16* Wuy1Eb = (bf16*)alloc(WB);
  bf16* Wuz2Eb = (bf16*)alloc(WB);
  bf16* WuDb   = (bf16*)alloc(WB);
  bf16* Wuy1Db = (bf16*)alloc(WB);
  bf16* Wuz2Db = (bf16*)alloc(WB);
  float* wzspe = (float*)alloc(128 * sizeof(float));
  float* wzspd = (float*)alloc(128 * sizeof(float));

  // Phase 1: encoders
  fnf_kernel<<<B_TOT, 128, 0, stream>>>(E, nullptr, 1, P1, WcRe1, WcIm1, Wp1,
                                        Wf1_1, bf1_1, Wf2_1, bf2_1, m1);
  fnf_kernel<<<B_TOT, 128, 0, stream>>>(Yg, edot0, 2, P2, WcRe2, WcIm2, Wp2,
                                        Wf1_2, bf1_2, Wf2_2, bf2_2, m2);
  // Phase 2: weight conversion + time-invariant PICNN context
  wconv_kernel<<<64, 256, 0, stream>>>(Wz2E, Wz2D, wzoutE, wzoutD,
                                       WuE, Wuy1E, Wuz2E, WuD, Wuy1D, Wuz2D,
                                       WspE, WspET, WspD, WspDT,
                                       WuEb, Wuy1Eb, Wuz2Eb, WuDb, Wuy1Db, Wuz2Db,
                                       wzspe, wzspd);
  gemm128_kernel<<<B_TOT / 16, 256, 0, stream>>>(m1,   WuEb,   nullptr, utmp, 1);
  gemm128_kernel<<<B_TOT / 16, 256, 0, stream>>>(utmp, Wuy1Eb, nullptr, c1e,  0);
  gemm128_kernel<<<B_TOT / 16, 256, 0, stream>>>(utmp, Wuz2Eb, b2E,     c2e,  0);
  gemm128_kernel<<<B_TOT / 16, 256, 0, stream>>>(m2,   WuDb,   nullptr, utmp, 1);
  gemm128_kernel<<<B_TOT / 16, 256, 0, stream>>>(utmp, Wuy1Db, nullptr, c1d,  0);
  gemm128_kernel<<<B_TOT / 16, 256, 0, stream>>>(utmp, Wuz2Db, b2D,     c2d,  0);
  // Phase 3: serial scan
  float* stress = (float*)d_out;
  float* xis    = (float*)d_out + (size_t)B_TOT * T_STEPS;
  timeloop_kernel<<<B_TOT / 16, 256, 0, stream>>>(
      e, c1e, c2e, c1d, c2d, WspE, WspET, WspD, WspDT, wzspe, wzspd,
      Wy1E, Wy2E, wyoutE, Wy1D, Wy2D, wyoutD, stress, xis);
}
